// GNNMLPv4_25228637896957
// MI455X (gfx1250) — matrix operand order repaired and checked by reading, not compiled, not run on hardware
//
#include <hip/hip_runtime.h>
#include <hip/hip_bf16.h>

#define NNODES  100000
#define NEDGES  600000
#define GDIM    128
#define NGRAPHS 8
#define NHEADS  3
#define NROWPAD 100096   // NNODES rounded up to 128-row GEMM tile

typedef __attribute__((ext_vector_type(16))) __bf16 v16bf;
typedef __attribute__((ext_vector_type(8)))  float  v8f;

__device__ __forceinline__ unsigned short f2bf(float f) {
  unsigned u = __float_as_uint(f);
  unsigned r = u + 0x7FFFu + ((u >> 16) & 1u);   // round-to-nearest-even
  return (unsigned short)(r >> 16);
}

__device__ __forceinline__ unsigned fkey(float f) {   // monotone float -> uint
  unsigned u = __float_as_uint(f);
  return (u & 0x80000000u) ? ~u : (u | 0x80000000u);
}

// ---------------------------------------------------------------------------
// Rearrange a 128x128 f32 weight matrix (row-major W[k][j], out = in @ W) into
// bf16 WMMA B-fragments: dst[((kk*8+nb)*32 + lane)*16 + e]
// lane<16 : j = nb*16+lane,    K {0..7,16..23} of the k-step
// lane>=16: j = nb*16+lane-16, K {8..15,24..31} of the k-step
// ---------------------------------------------------------------------------
__global__ __launch_bounds__(256) void prep_wfrag(const float* __restrict__ W,
                                                  unsigned short* __restrict__ dst) {
  int t = blockIdx.x * 256 + threadIdx.x;      // 0..16383
  int e    = t & 15;
  int lane = (t >> 4) & 31;
  int nb   = (t >> 9) & 7;
  int kk   = t >> 12;
  int l16  = lane & 15;
  int hi   = lane >> 4;
  int j = nb * 16 + l16;
  int k = kk * 32 + (e < 8 ? hi * 8 + e : 16 + hi * 8 + (e - 8));
  dst[t] = f2bf(W[k * GDIM + j]);
}

// Fold bias + eval-mode BatchNorm into alpha/beta: out = relu(gemm*alpha + beta)
__global__ void prep_ab(const float* __restrict__ g,  const float* __restrict__ be,
                        const float* __restrict__ rm, const float* __restrict__ rv,
                        const float* __restrict__ bias, float* __restrict__ ab) {
  int j = threadIdx.x;                         // 128 threads
  float al = g[j] * rsqrtf(rv[j] + 1e-5f);
  ab[j]        = al;
  ab[GDIM + j] = be[j] + (bias[j] - rm[j]) * al;
}

__global__ void init_wsi(unsigned* __restrict__ w) {
  if (threadIdx.x < NHEADS * NGRAPHS) w[threadIdx.x] = 0x007FFFFFu;  // fkey(-inf)
}

__global__ __launch_bounds__(256) void zero4(float4* __restrict__ p, int n4) {
  int t = blockIdx.x * 256 + threadIdx.x;
  if (t < n4) p[t] = make_float4(0.f, 0.f, 0.f, 0.f);
}

// ---------------------------------------------------------------------------
// out[m,:] = relu( ((in1[m,:] (+ in2[m,:])) @ W) * alpha + beta )
// 8 waves/block; wave w owns rows blockIdx*128 + 16w .. +15; 32 WMMAs/wave.
// Loads clamp the row index (always-valid reads of real inputs); stores are
// UNCONDITIONAL -- `out` must be padded to NROWPAD rows (workspace buffers).
// ---------------------------------------------------------------------------
__global__ __launch_bounds__(256) void gemm128_bn_relu(
    const float* __restrict__ in1, const float* __restrict__ in2,
    const unsigned short* __restrict__ wfrag, const float* __restrict__ ab,
    float* __restrict__ out, int nrows) {
  const int lane = threadIdx.x & 31;
  const int wave = threadIdx.x >> 5;
  const int l16  = lane & 15;
  const int hi   = lane >> 4;
  const int rowbase = blockIdx.x * 128 + wave * 16;
  int m = rowbase + l16;
  if (m >= nrows) m = nrows - 1;               // clamp: loads always valid
  const float4* r1 = (const float4*)(in1 + (size_t)m * GDIM);
  const float4* r2 = in2 ? (const float4*)(in2 + (size_t)m * GDIM) : nullptr;

  // A fragments (ISA 16-bit 16x32 layout): lanes<16 hold K {0..7,16..23},
  // lanes>=16 hold K {8..15,24..31} of each 32-wide k-step.
  v16bf afr[4];
#pragma unroll
  for (int kk = 0; kk < 4; ++kk) {
    const int q = kk * 8 + hi * 2;             // float4 index of K base
    float4 a0 = r1[q], a1 = r1[q + 1], a2 = r1[q + 4], a3 = r1[q + 5];
    if (r2) {
      float4 b0 = r2[q], b1 = r2[q + 1], b2 = r2[q + 4], b3 = r2[q + 5];
      a0.x += b0.x; a0.y += b0.y; a0.z += b0.z; a0.w += b0.w;
      a1.x += b1.x; a1.y += b1.y; a1.z += b1.z; a1.w += b1.w;
      a2.x += b2.x; a2.y += b2.y; a2.z += b2.z; a2.w += b2.w;
      a3.x += b3.x; a3.y += b3.y; a3.z += b3.z; a3.w += b3.w;
    }
    v16bf a;
    a[0]  = (__bf16)a0.x; a[1]  = (__bf16)a0.y; a[2]  = (__bf16)a0.z; a[3]  = (__bf16)a0.w;
    a[4]  = (__bf16)a1.x; a[5]  = (__bf16)a1.y; a[6]  = (__bf16)a1.z; a[7]  = (__bf16)a1.w;
    a[8]  = (__bf16)a2.x; a[9]  = (__bf16)a2.y; a[10] = (__bf16)a2.z; a[11] = (__bf16)a2.w;
    a[12] = (__bf16)a3.x; a[13] = (__bf16)a3.y; a[14] = (__bf16)a3.z; a[15] = (__bf16)a3.w;
    afr[kk] = a;
  }

  const v16bf* wf = (const v16bf*)wfrag;       // B fragments, L2-resident (32 KB)
#pragma unroll
  for (int nb = 0; nb < 8; ++nb) {
    v8f c = {};
#pragma unroll
    for (int kk = 0; kk < 4; ++kk) {
      v16bf b = wf[(kk * 8 + nb) * 32 + lane];
      c = __builtin_amdgcn_wmma_f32_16x16x32_bf16(false, afr[kk], false, b,
                                                  (short)0, c, false, false);
    }
    const int   jj = nb * 16 + l16;
    const float al = ab[jj];
    const float bt = ab[GDIM + jj];
    float* op = out + (size_t)(rowbase + hi * 8) * GDIM + jj;  // C: hi lanes = M 8..15
#pragma unroll
    for (int r = 0; r < 8; ++r) {              // unconditional, padded buffer
      float v = c[r] * al + bt;
      op[(size_t)r * GDIM] = v > 0.f ? v : 0.f;
    }
  }
}

// One wave per edge: lane covers 4 consecutive dims; atomic f32 adds hit L2.
__global__ __launch_bounds__(256) void scatter_add_edges(
    const float* __restrict__ h, const int* __restrict__ src,
    const int* __restrict__ dst, float* __restrict__ agg, int nedges) {
  int t = blockIdx.x * 256 + threadIdx.x;
  int e = t >> 5, lane = t & 31;
  if (e >= nedges) return;
  int s = src[e], d = dst[e];
  const float4 v = *(const float4*)(h + (size_t)s * GDIM + lane * 4);
  float* ap = agg + (size_t)d * GDIM + lane * 4;
  atomicAdd(ap + 0, v.x); atomicAdd(ap + 1, v.y);
  atomicAdd(ap + 2, v.z); atomicAdd(ap + 3, v.w);
}

// node = z2 @ w3 + b3 (128->1); sigmoid to out; run-length-compressed
// segment-max (batch sorted) via atomicMax on monotone key, pre-sigmoid.
__global__ __launch_bounds__(256) void head_final(
    const float* __restrict__ z2, const float* __restrict__ w3,
    const float* __restrict__ b3p, const int* __restrict__ batch,
    float* __restrict__ node_out, unsigned* __restrict__ wsi_raw, int nrows) {
  const int lane = threadIdx.x & 31;
  const int wave = threadIdx.x >> 5;
  const float4 w = *(const float4*)(w3 + lane * 4);
  const float  b3 = b3p[0];
  int n0 = (blockIdx.x * 8 + wave) * 32;       // 32 nodes per wave
  int gcur = -1; float gmax = 0.f;
  for (int i = 0; i < 32; ++i) {
    int n = n0 + i;
    if (n >= nrows) break;
    const float4 z = *(const float4*)(z2 + (size_t)n * GDIM + lane * 4);
    float p = z.x * w.x + z.y * w.y + z.z * w.z + z.w * w.w;
#pragma unroll
    for (int off = 16; off > 0; off >>= 1) p += __shfl_xor(p, off, 32);
    if (lane == 0) {
      float val = p + b3;
      node_out[n] = 1.0f / (1.0f + __expf(-val));
      int g = batch[n];
      if (g != gcur) {
        if (gcur >= 0) atomicMax(&wsi_raw[gcur], fkey(gmax));
        gcur = g; gmax = val;
      } else if (val > gmax) gmax = val;
    }
  }
  if (lane == 0 && gcur >= 0) atomicMax(&wsi_raw[gcur], fkey(gmax));
}

__global__ void finalize_wsi(const unsigned* __restrict__ wsi_raw,
                             float* __restrict__ out) {
  int i = threadIdx.x;
  if (i < NHEADS * NGRAPHS) {
    unsigned k = wsi_raw[i];
    float v = __uint_as_float((k & 0x80000000u) ? (k & 0x7FFFFFFFu) : ~k);
    out[i] = 1.0f / (1.0f + __expf(-v));
  }
}

extern "C" void kernel_launch(void* const* d_in, const int* in_sizes, int n_in,
                              void* d_out, int out_size, void* d_ws, size_t ws_size,
                              hipStream_t stream) {
  const float* x     = (const float*)d_in[0];
  const int*   src   = (const int*)d_in[1];
  const int*   dst   = src + NEDGES;
  const int*   batch = (const int*)d_in[2];
  const float* W_first = (const float*)d_in[3];
  const float* b_first = (const float*)d_in[4];
  const float* g_first = (const float*)d_in[5];
  const float* be_first = (const float*)d_in[6];
  const float* rm_first = (const float*)d_in[7];
  const float* rv_first = (const float*)d_in[8];
  const float* W_conv = (const float*)d_in[9];
  const float* b_conv = (const float*)d_in[10];
  const float* g_conv = (const float*)d_in[11];
  const float* be_conv = (const float*)d_in[12];
  const float* rm_conv = (const float*)d_in[13];
  const float* rv_conv = (const float*)d_in[14];
  const float* W1 = (const float*)d_in[15];
  const float* b1 = (const float*)d_in[16];
  const float* g1 = (const float*)d_in[17];
  const float* be1 = (const float*)d_in[18];
  const float* rm1 = (const float*)d_in[19];
  const float* rv1 = (const float*)d_in[20];
  const float* W2 = (const float*)d_in[21];
  const float* b2 = (const float*)d_in[22];
  const float* g2 = (const float*)d_in[23];
  const float* be2 = (const float*)d_in[24];
  const float* rm2 = (const float*)d_in[25];
  const float* rv2 = (const float*)d_in[26];
  const float* W3 = (const float*)d_in[27];
  const float* b3 = (const float*)d_in[28];

  // workspace partition (activation buffers padded to the 128-row tile)
  char* ws = (char*)d_ws;
  const size_t matbytes = (size_t)NROWPAD * GDIM * sizeof(float);  // ~51.25 MB
  float* buf0 = (float*)ws; ws += matbytes;
  float* buf1 = (float*)ws; ws += matbytes;
  float* agg  = (float*)ws; ws += matbytes;
  unsigned short* wfrag = (unsigned short*)ws; ws += (size_t)9 * GDIM * GDIM * sizeof(unsigned short);
  float* ab = (float*)ws; ws += (size_t)9 * 2 * GDIM * sizeof(float);
  unsigned* wsi_raw = (unsigned*)ws; ws += 256;

  float* out      = (float*)d_out;
  float* out_wsi  = out;                       // [3,8,1]
  float* out_node = out + NHEADS * NGRAPHS;    // [3,100000,1]

  // --- weight prep (sets: 0=first, 1..2=conv, 3..5=W1[h], 6..8=W2[h]) ---
  const float* Wsrc[9] = { W_first, W_conv, W_conv + GDIM * GDIM,
                           W1, W1 + GDIM * GDIM, W1 + 2 * GDIM * GDIM,
                           W2, W2 + GDIM * GDIM, W2 + 2 * GDIM * GDIM };
  for (int i = 0; i < 9; ++i)
    prep_wfrag<<<64, 256, 0, stream>>>(Wsrc[i], wfrag + (size_t)i * GDIM * GDIM);

  prep_ab<<<1, 128, 0, stream>>>(g_first, be_first, rm_first, rv_first, b_first, ab);
  for (int l = 0; l < 2; ++l)
    prep_ab<<<1, 128, 0, stream>>>(g_conv + l * GDIM, be_conv + l * GDIM,
                                   rm_conv + l * GDIM, rv_conv + l * GDIM,
                                   b_conv + l * GDIM, ab + (1 + l) * 2 * GDIM);
  for (int h = 0; h < 3; ++h)
    prep_ab<<<1, 128, 0, stream>>>(g1 + h * GDIM, be1 + h * GDIM, rm1 + h * GDIM,
                                   rv1 + h * GDIM, b1 + h * GDIM, ab + (3 + h) * 2 * GDIM);
  for (int h = 0; h < 3; ++h)
    prep_ab<<<1, 128, 0, stream>>>(g2 + h * GDIM, be2 + h * GDIM, rm2 + h * GDIM,
                                   rv2 + h * GDIM, b2 + h * GDIM, ab + (6 + h) * 2 * GDIM);
  init_wsi<<<1, 32, 0, stream>>>(wsi_raw);

  const int GGRID = (NNODES + 127) / 128;                 // 782
  const int ZGRID = (NNODES * GDIM / 4 + 255) / 256;      // 12500
  const int SGRID = (NEDGES * 32 + 255) / 256;            // 75000
  const int HGRID = (NNODES + 255) / 256;                 // 391

  // layer 0
  gemm128_bn_relu<<<GGRID, 256, 0, stream>>>(x, nullptr, wfrag, ab, buf0, NNODES);
  // GIN conv 0
  zero4<<<ZGRID, 256, 0, stream>>>((float4*)agg, NNODES * GDIM / 4);
  scatter_add_edges<<<SGRID, 256, 0, stream>>>(buf0, src, dst, agg, NEDGES);
  gemm128_bn_relu<<<GGRID, 256, 0, stream>>>(buf0, agg, wfrag + (size_t)1 * GDIM * GDIM,
                                             ab + 1 * 2 * GDIM, buf1, NNODES);
  // GIN conv 1
  zero4<<<ZGRID, 256, 0, stream>>>((float4*)agg, NNODES * GDIM / 4);
  scatter_add_edges<<<SGRID, 256, 0, stream>>>(buf1, src, dst, agg, NEDGES);
  gemm128_bn_relu<<<GGRID, 256, 0, stream>>>(buf1, agg, wfrag + (size_t)2 * GDIM * GDIM,
                                             ab + 2 * 2 * GDIM, buf0, NNODES);
  // heads
  for (int h = 0; h < 3; ++h) {
    gemm128_bn_relu<<<GGRID, 256, 0, stream>>>(buf0, nullptr,
        wfrag + (size_t)(3 + h) * GDIM * GDIM, ab + (3 + h) * 2 * GDIM, buf1, NNODES);
    gemm128_bn_relu<<<GGRID, 256, 0, stream>>>(buf1, nullptr,
        wfrag + (size_t)(6 + h) * GDIM * GDIM, ab + (6 + h) * 2 * GDIM, agg, NNODES);
    head_final<<<HGRID, 256, 0, stream>>>(agg, W3 + h * GDIM, b3 + h, batch,
                                          out_node + (size_t)h * NNODES,
                                          wsi_raw + h * NGRAPHS, NNODES);
  }
  finalize_wsi<<<1, 32, 0, stream>>>(wsi_raw, out_wsi);
}
